// ImprovedGAT_37022618092007
// MI455X (gfx1250) — matrix operand order repaired and checked by reading, not compiled, not run on hardware
//
#include <hip/hip_runtime.h>
#include <math.h>

// ---------------- problem constants ----------------
#define NN   50000
#define NE   200000
#define NG   500
#define DIN  5
#define HID  128
#define HD1  6
#define HD2  4
#define OUTC 7

typedef __attribute__((ext_vector_type(16))) __bf16 v16bf;
typedef __attribute__((ext_vector_type(8)))  __bf16 v8bf;
typedef __attribute__((ext_vector_type(4)))  __bf16 v4bf;
typedef __attribute__((ext_vector_type(8)))  float  v8f;

// order-preserving float <-> uint encoding for atomicMax-based segment max
__device__ __forceinline__ unsigned enc_f(float f) {
  unsigned u = __float_as_uint(f);
  return (u & 0x80000000u) ? ~u : (u | 0x80000000u);
}
__device__ __forceinline__ float dec_f(unsigned u) {
  return (u & 0x80000000u) ? __uint_as_float(u & 0x7FFFFFFFu)
                           : __uint_as_float(~u);
}
#define ENC_NEG_FLT_MAX 0x00800000u   // enc(-FLT_MAX)

// ---------------- WMMA GEMM: C[M,N] = A[M,K] @ W[K,N] (+bias) (+=C) ----------------
// block = 256 threads = 8 waves; block tile 64(M) x 64(N); K step 32.
// Each wave computes 16x32 (two 16x16 accumulators sharing the A fragment).
// A staged row-major As[64][40] (bf16), W staged transposed Bt[64][40] so both
// fragments are contiguous per lane -> ds_load_b128. Interior tiles use float4
// global loads; edge tiles fall back to bounds-checked scalar loads.
__global__ __launch_bounds__(256) void wmma_gemm_k(
    const float* __restrict__ A, const float* __restrict__ W,
    const float* __restrict__ bias, float* __restrict__ C,
    int M, int Ncols, int K, int accum)
{
  __shared__ __align__(16) __bf16 As[64][40];   // [m][k], row stride 80B (16B mult)
  __shared__ __align__(16) __bf16 Bt[64][40];   // [n][k], transposed W tile
  const int tid  = threadIdx.x;
  const int lane = tid & 31;
  const int wv   = tid >> 5;
  const int wm   = wv >> 1;           // 0..3  (M direction, 16 rows each)
  const int wn   = wv & 1;            // 0..1  (N direction, 32 cols each)
  const int half = lane >> 4;
  const int loc  = lane & 15;
  const int mBase = blockIdx.y * 64;
  const int nBase = blockIdx.x * 64;

  const bool intM = (mBase + 64 <= M);
  const bool intN = (nBase + 64 <= Ncols);

  v8f acc0 = {};
  v8f acc1 = {};
  for (int k0 = 0; k0 < K; k0 += 32) {
    const bool intK = (k0 + 32 <= K);

    // ---- stage A tile 64x32 ----
    if (intM && intK) {
      // 512 float4s, 2 per thread; global_load_b128 + ds_store_b64
#pragma unroll
      for (int t = 0; t < 2; ++t) {
        int idx = tid + 256 * t;
        int r  = idx >> 3;            // 8 float4 per 32-wide row
        int c4 = idx & 7;
        float4 v = ((const float4*)(A + (size_t)(mBase + r) * K + k0))[c4];
        v4bf b;
        b[0] = (__bf16)v.x; b[1] = (__bf16)v.y; b[2] = (__bf16)v.z; b[3] = (__bf16)v.w;
        *(v4bf*)(&As[r][c4 * 4]) = b;
      }
    } else {
      for (int i = tid; i < 64 * 32; i += 256) {
        int r = i >> 5, c = i & 31;
        int gm = mBase + r, gk = k0 + c;
        float v = (gm < M && gk < K) ? A[(size_t)gm * K + gk] : 0.0f;
        As[r][c] = (__bf16)v;
      }
    }

    // ---- stage W tile 32x64 (transposed into Bt[n][k]) ----
    if (intN && intK) {
#pragma unroll
      for (int t = 0; t < 2; ++t) {
        int idx = tid + 256 * t;
        int r  = idx >> 4;            // k row, 16 float4 per 64-wide row
        int c4 = idx & 15;
        float4 v = ((const float4*)(W + (size_t)(k0 + r) * Ncols + nBase))[c4];
        int n0 = c4 * 4;
        Bt[n0 + 0][r] = (__bf16)v.x;
        Bt[n0 + 1][r] = (__bf16)v.y;
        Bt[n0 + 2][r] = (__bf16)v.z;
        Bt[n0 + 3][r] = (__bf16)v.w;
      }
    } else {
      for (int i = tid; i < 32 * 64; i += 256) {
        int r = i >> 6, c = i & 63;   // r = k, c = n
        int gk = k0 + r, gn = nBase + c;
        float v = (gk < K && gn < Ncols) ? W[(size_t)gk * Ncols + gn] : 0.0f;
        Bt[c][r] = (__bf16)v;
      }
    }
    __syncthreads();

    // ---- fragments (contiguous per lane -> wide ds loads) ----
    // A 16x32 bf16 layout: lane holds row M=loc; regs pack K = 16*half..16*half+15
    v16bf af = *(const v16bf*)(&As[wm * 16 + loc][16 * half]);
    // B 32x16 bf16 layout: lane holds col N; regs pack K = 16*half..16*half+15
    v16bf b0 = *(const v16bf*)(&Bt[wn * 32 + loc][16 * half]);
    v16bf b1 = *(const v16bf*)(&Bt[wn * 32 + 16 + loc][16 * half]);

    acc0 = __builtin_amdgcn_wmma_f32_16x16x32_bf16(false, af, false, b0,
                                                   (short)0, acc0, false, false);
    acc1 = __builtin_amdgcn_wmma_f32_16x16x32_bf16(false, af, false, b1,
                                                   (short)0, acc1, false, false);
    __syncthreads();
  }

#pragma unroll
  for (int r = 0; r < 8; ++r) {
    int row  = mBase + wm * 16 + r + 8 * half;   // C layout: VGPR r -> M = r + 8*half
    int col0 = nBase + wn * 32 + loc;            // N = lane&15 (+16 for acc1)
    if (row < M) {
      if (col0 < Ncols) {
        float v = acc0[r];
        if (bias) v += bias[col0];
        size_t o = (size_t)row * Ncols + col0;
        if (accum) v += C[o];
        C[o] = v;
      }
      int col1 = col0 + 16;
      if (col1 < Ncols) {
        float v = acc1[r];
        if (bias) v += bias[col1];
        size_t o = (size_t)row * Ncols + col1;
        if (accum) v += C[o];
        C[o] = v;
      }
    }
  }
}

// ---------------- utility kernels ----------------
__global__ void fill_u32_k(unsigned* p, unsigned v, long long n) {
  long long i = (long long)blockIdx.x * blockDim.x + threadIdx.x;
  if (i < n) p[i] = v;
}

__global__ void rows_bias_k(float* out, const float* __restrict__ bias,
                            long long total, int Nc) {
  long long i = (long long)blockIdx.x * blockDim.x + threadIdx.x;
  if (i < total) out[i] = bias[i % Nc];
}

__global__ void col_stats_k(const float* __restrict__ X, long long M, int Nc,
                            float* s, float* s2) {
  int c = blockIdx.x * blockDim.x + threadIdx.x;
  if (c >= Nc) return;
  long long rowsPer = (M + gridDim.y - 1) / gridDim.y;
  long long r0 = (long long)blockIdx.y * rowsPer;
  long long r1 = r0 + rowsPer; if (r1 > M) r1 = M;
  float a = 0.f, b = 0.f;
  for (long long r = r0; r < r1; ++r) {
    float v = X[r * Nc + c];
    a += v; b += v * v;
  }
  atomicAdd(&s[c], a);
  atomicAdd(&s2[c], b);
}

__global__ void bn_fin_k(const float* s, const float* s2, const float* g,
                         const float* be, float M, int Nc,
                         float* scale, float* shift) {
  int c = blockIdx.x * blockDim.x + threadIdx.x;
  if (c >= Nc) return;
  float mean = s[c] / M;
  float var  = fmaxf(s2[c] / M - mean * mean, 0.0f);
  float inv  = rsqrtf(var + 1e-5f);
  float sc   = g[c] * inv;
  scale[c] = sc;
  shift[c] = be[c] - mean * sc;
}

__global__ void bn_lrelu_k(float* X, const float* __restrict__ scale,
                           const float* __restrict__ shift,
                           long long total, int Nc, float slope) {
  long long i = (long long)blockIdx.x * blockDim.x + threadIdx.x;
  if (i >= total) return;
  int c = (int)(i % Nc);
  float v = X[i] * scale[c] + shift[c];
  X[i] = v >= 0.f ? v : slope * v;
}

__global__ void lrelu_k(float* X, long long n, float slope) {
  long long i = (long long)blockIdx.x * blockDim.x + threadIdx.x;
  if (i >= n) return;
  float v = X[i];
  X[i] = v >= 0.f ? v : slope * v;
}

// ---------------- GAT attention ----------------
__global__ void attn_logits_k(const float* __restrict__ hh,
                              const float* __restrict__ a_s,
                              const float* __restrict__ a_d,
                              float* als, float* ald, int n, int heads) {
  long long idx = (long long)blockIdx.x * blockDim.x + threadIdx.x;
  long long total = (long long)n * heads;
  if (idx >= total) return;
  int node = (int)(idx / heads), h = (int)(idx % heads);
  const float* hv  = hh  + ((size_t)node * heads + h) * HID;
  const float* avs = a_s + (size_t)h * HID;
  const float* avd = a_d + (size_t)h * HID;
  float ss = 0.f, sd = 0.f;
  for (int c = 0; c < HID; ++c) { float v = hv[c]; ss += v * avs[c]; sd += v * avd[c]; }
  als[idx] = ss; ald[idx] = sd;
}

__global__ void edge_max_k(const float* __restrict__ als,
                           const float* __restrict__ ald,
                           const int* __restrict__ src, const int* __restrict__ dst,
                           int E, int n, int heads, float* ebuf, unsigned* nmax) {
  long long idx = (long long)blockIdx.x * blockDim.x + threadIdx.x;
  long long total = (long long)(E + n) * heads;
  if (idx >= total) return;
  int e = (int)(idx / heads), h = (int)(idx % heads);
  int s = (e < E) ? src[e] : (e - E);
  int d = (e < E) ? dst[e] : (e - E);
  float v = als[(size_t)s * heads + h] + ald[(size_t)d * heads + h];
  v = v >= 0.f ? v : 0.2f * v;                 // attention LeakyReLU(0.2)
  ebuf[idx] = v;
  atomicMax(&nmax[(size_t)d * heads + h], enc_f(v));
}

__global__ void edge_exp_k(float* ebuf, const unsigned* __restrict__ nmax,
                           const int* __restrict__ dst, int E, int n, int heads,
                           float* nsum) {
  long long idx = (long long)blockIdx.x * blockDim.x + threadIdx.x;
  long long total = (long long)(E + n) * heads;
  if (idx >= total) return;
  int e = (int)(idx / heads), h = (int)(idx % heads);
  int d = (e < E) ? dst[e] : (e - E);
  float m = dec_f(nmax[(size_t)d * heads + h]);  // self-loops -> always valid
  float ex = expf(ebuf[idx] - m);
  ebuf[idx] = ex;
  atomicAdd(&nsum[(size_t)d * heads + h], ex);
}

__global__ void edge_scatter_k(const float* __restrict__ hh,
                               const float* __restrict__ ebuf,
                               const float* __restrict__ nsum,
                               const int* __restrict__ src, const int* __restrict__ dst,
                               int E, int n, int heads, float* out) {
  long long idx = (long long)blockIdx.x * blockDim.x + threadIdx.x;
  long long total = (long long)(E + n) * heads;
  if (idx >= total) return;
  int e = (int)(idx / heads), h = (int)(idx % heads);
  int s = (e < E) ? src[e] : (e - E);
  int d = (e < E) ? dst[e] : (e - E);
  float w = ebuf[idx] / fmaxf(nsum[(size_t)d * heads + h], 1e-16f);
  const float* hs = hh  + ((size_t)s * heads + h) * HID;
  float*       od = out + ((size_t)d * heads + h) * HID;
  for (int c = 0; c < HID; ++c) atomicAdd(&od[c], w * hs[c]);
}

// ---------------- GCN ----------------
__global__ void deg_k(const int* __restrict__ dst, float* deg, int E) {
  int i = blockIdx.x * blockDim.x + threadIdx.x;
  if (i < E) atomicAdd(&deg[dst[i]], 1.0f);
}
__global__ void dis_k(float* deg, int n) {
  int i = blockIdx.x * blockDim.x + threadIdx.x;
  if (i < n) deg[i] = rsqrtf(deg[i]);          // deg >= 1 due to self-loop
}
__global__ void gcn_init_k(float* out, const float* __restrict__ hw,
                           const float* __restrict__ dis,
                           const float* __restrict__ bias,
                           long long total, int C) {
  long long i = (long long)blockIdx.x * blockDim.x + threadIdx.x;
  if (i >= total) return;
  int n = (int)(i / C), c = (int)(i % C);
  float d = dis[n];
  out[i] = bias[c] + d * d * hw[i];            // self-loop contribution
}
__global__ void gcn_scatter_k(const float* __restrict__ hw,
                              const float* __restrict__ dis,
                              const int* __restrict__ src, const int* __restrict__ dst,
                              float* out, int E, int C) {
  int i = blockIdx.x * blockDim.x + threadIdx.x;
  if (i >= E) return;
  int s = src[i], d = dst[i];
  float nr = dis[s] * dis[d];
  const float* hs = hw + (size_t)s * C;
  float*       od = out + (size_t)d * C;
  for (int c = 0; c < C; ++c) atomicAdd(&od[c], nr * hs[c]);
}

// ---------------- gated fusion ----------------
__global__ void gate_combine_k(float* gate_xc, const float* __restrict__ xg2,
                               const float* __restrict__ xgc, long long total) {
  long long i = (long long)blockIdx.x * blockDim.x + threadIdx.x;
  if (i >= total) return;
  float g = 1.0f / (1.0f + expf(-gate_xc[i]));
  gate_xc[i] = g * xg2[i] + (1.0f - g) * xgc[i];   // overwrite gate with xc
}

// ---------------- attention pooling ----------------
__global__ void score_k(const float* __restrict__ T, const float* __restrict__ Wp2,
                        const float* __restrict__ bp2, float* sc, int n) {
  int i = blockIdx.x * blockDim.x + threadIdx.x;
  if (i >= n) return;
  float s = bp2[0];
  const float* tv = T + (size_t)i * HID;
  for (int c = 0; c < HID; ++c) s += tv[c] * Wp2[c];
  sc[i] = s;
}
__global__ void seg_max_k(const float* __restrict__ sc, const int* __restrict__ seg,
                          unsigned* gm, int n) {
  int i = blockIdx.x * blockDim.x + threadIdx.x;
  if (i < n) atomicMax(&gm[seg[i]], enc_f(sc[i]));
}
__global__ void seg_exp_k(float* sc, const int* __restrict__ seg,
                          const unsigned* __restrict__ gm, float* gs, int n) {
  int i = blockIdx.x * blockDim.x + threadIdx.x;
  if (i >= n) return;
  float m = dec_f(gm[seg[i]]);
  if (m < -3.0e38f) m = 0.f;                   // empty-segment guard (isfinite)
  float e = expf(sc[i] - m);
  sc[i] = e;
  atomicAdd(&gs[seg[i]], e);
}
__global__ void pool_scatter_k(const float* __restrict__ sc,
                               const float* __restrict__ gs,
                               const int* __restrict__ seg,
                               const float* __restrict__ xc,
                               float* pl, int n, int C) {
  int i = blockIdx.x * blockDim.x + threadIdx.x;
  if (i >= n) return;
  int g = seg[i];
  float w = sc[i] / fmaxf(gs[g], 1e-16f);
  const float* xv = xc + (size_t)i * C;
  float*       ov = pl + (size_t)g * C;
  for (int c = 0; c < C; ++c) atomicAdd(&ov[c], w * xv[c]);
}

// ---------------- final log-softmax ----------------
__global__ void log_softmax_k(const float* __restrict__ L, float* out, int G, int C) {
  int g = blockIdx.x * blockDim.x + threadIdx.x;
  if (g >= G) return;
  float m = -3.4e38f;
  for (int j = 0; j < C; ++j) m = fmaxf(m, L[(size_t)g * C + j]);
  float s = 0.f;
  for (int j = 0; j < C; ++j) s += expf(L[(size_t)g * C + j] - m);
  float ls = logf(s);
  for (int j = 0; j < C; ++j) out[(size_t)g * C + j] = L[(size_t)g * C + j] - m - ls;
}

// ---------------- host orchestration ----------------
static inline unsigned cdiv(long long a, long long b) { return (unsigned)((a + b - 1) / b); }

extern "C" void kernel_launch(void* const* d_in, const int* in_sizes, int n_in,
                              void* d_out, int out_size, void* d_ws, size_t ws_size,
                              hipStream_t stream) {
  (void)in_sizes; (void)n_in; (void)out_size; (void)ws_size;

  const float* x     = (const float*)d_in[0];
  const int*   ei    = (const int*)d_in[1];
  const int*   bat   = (const int*)d_in[2];
  const float* W_in  = (const float*)d_in[3];
  const float* b_in  = (const float*)d_in[4];
  const float* g_in  = (const float*)d_in[5];
  const float* be_in = (const float*)d_in[6];
  const float* W1    = (const float*)d_in[7];
  const float* a1s   = (const float*)d_in[8];
  const float* a1d   = (const float*)d_in[9];
  const float* b1    = (const float*)d_in[10];
  const float* g1    = (const float*)d_in[11];
  const float* be1   = (const float*)d_in[12];
  const float* W2    = (const float*)d_in[13];
  const float* a2s   = (const float*)d_in[14];
  const float* a2d   = (const float*)d_in[15];
  const float* b2    = (const float*)d_in[16];
  const float* g2    = (const float*)d_in[17];
  const float* be2   = (const float*)d_in[18];
  const float* Wg    = (const float*)d_in[19];
  const float* bg    = (const float*)d_in[20];
  const float* gg    = (const float*)d_in[21];
  const float* beg   = (const float*)d_in[22];
  const float* Wgate = (const float*)d_in[23];
  const float* bgate = (const float*)d_in[24];
  const float* Wp1   = (const float*)d_in[25];
  const float* bp1   = (const float*)d_in[26];
  const float* Wp2   = (const float*)d_in[27];
  const float* bp2   = (const float*)d_in[28];
  const float* Wc1   = (const float*)d_in[29];
  const float* bc1   = (const float*)d_in[30];
  const float* gc    = (const float*)d_in[31];
  const float* bec   = (const float*)d_in[32];
  const float* Wc2   = (const float*)d_in[33];
  const float* bc2   = (const float*)d_in[34];

  const int* src = ei;
  const int* dst = ei + NE;
  const long long ET = NE + NN;

  // workspace layout (floats)
  float* wf = (float*)d_ws;
  const size_t OFF_H   = 0;                                  // h            [N,128]
  const size_t OFF_A   = OFF_H  + (size_t)NN * HID;          // hh1/hh2/xgc  [N,768]
  const size_t OFF_B   = OFF_A  + (size_t)NN * HID * HD1;    // xg1/hwg/xc   [N,768]
  const size_t OFF_C   = OFF_B  + (size_t)NN * HID * HD1;    // xg2          [N,512]
  const size_t OFF_E   = OFF_C  + (size_t)NN * HID * HD2;    // edge buf     [ET,6]
  const size_t OFF_NM  = OFF_E  + (size_t)ET * HD1;          // node max     [N,6]
  const size_t OFF_NS  = OFF_NM + (size_t)NN * HD1;          // node sum     [N,6]
  const size_t OFF_ALS = OFF_NS + (size_t)NN * HD1;          // alpha src    [N,6]
  const size_t OFF_ALD = OFF_ALS + (size_t)NN * HD1;         // alpha dst    [N,6]
  const size_t OFF_DEG = OFF_ALD + (size_t)NN * HD1;         // deg/dis      [N]
  const size_t OFF_SC  = OFF_DEG + (size_t)NN;               // node score   [N]
  const size_t OFF_GM  = OFF_SC  + (size_t)NN;               // graph max    [G]
  const size_t OFF_GS  = OFF_GM  + (size_t)NG;               // graph sum    [G]
  const size_t OFF_PL  = OFF_GS  + (size_t)NG;               // pooled       [G,512]
  const size_t OFF_HC  = OFF_PL  + (size_t)NG * 512;         // hcl          [G,128]
  const size_t OFF_LG  = OFF_HC  + (size_t)NG * HID;         // logits       [G,8]
  const size_t OFF_S1  = OFF_LG  + (size_t)NG * 8;           // col sums     [1024]
  const size_t OFF_S2  = OFF_S1  + 1024;
  const size_t OFF_SCL = OFF_S2  + 1024;
  const size_t OFF_SHF = OFF_SCL + 1024;

  float* H   = wf + OFF_H;   float* A   = wf + OFF_A;   float* B   = wf + OFF_B;
  float* Cb  = wf + OFF_C;   float* EB  = wf + OFF_E;
  unsigned* NM = (unsigned*)(wf + OFF_NM);
  float* NS  = wf + OFF_NS;  float* ALS = wf + OFF_ALS; float* ALD = wf + OFF_ALD;
  float* DEG = wf + OFF_DEG; float* SC  = wf + OFF_SC;
  unsigned* GM = (unsigned*)(wf + OFF_GM);
  float* GS  = wf + OFF_GS;  float* PL  = wf + OFF_PL;  float* HC  = wf + OFF_HC;
  float* LG  = wf + OFF_LG;
  float* S1  = wf + OFF_S1;  float* S2  = wf + OFF_S2;
  float* SCL = wf + OFF_SCL; float* SHF = wf + OFF_SHF;

  const int T = 256;
  #define GEMM(Ain, Win, bia, Cout, M_, N_, K_, acc_) \
    wmma_gemm_k<<<dim3(cdiv((N_),64), cdiv((M_),64)), 256, 0, stream>>>((Ain),(Win),(bia),(Cout),(M_),(N_),(K_),(acc_))
  #define STATS(X_, M_, Nc_, Gm_, Bm_) do {                                            \
    fill_u32_k<<<cdiv(2048,T),T,0,stream>>>((unsigned*)S1, 0u, 2048);                  \
    col_stats_k<<<dim3(cdiv((Nc_),T),128),T,0,stream>>>((X_),(M_),(Nc_),S1,S2);        \
    bn_fin_k<<<cdiv((Nc_),T),T,0,stream>>>(S1,S2,(Gm_),(Bm_),(float)(M_),(Nc_),SCL,SHF); } while(0)

  // ---- input linear + BN + lrelu(0.01) -> h ----
  GEMM(x, W_in, b_in, H, NN, HID, DIN, 0);
  STATS(H, NN, HID, g_in, be_in);
  bn_lrelu_k<<<cdiv((long long)NN*HID,T),T,0,stream>>>(H,SCL,SHF,(long long)NN*HID,HID,0.01f);

  // ---- GAT layer 1 (6 heads) ----
  GEMM(H, W1, (const float*)nullptr, A, NN, HID*HD1, HID, 0);           // hh1
  attn_logits_k<<<cdiv((long long)NN*HD1,T),T,0,stream>>>(A,a1s,a1d,ALS,ALD,NN,HD1);
  fill_u32_k<<<cdiv((long long)NN*HD1,T),T,0,stream>>>(NM,ENC_NEG_FLT_MAX,(long long)NN*HD1);
  edge_max_k<<<cdiv(ET*HD1,T),T,0,stream>>>(ALS,ALD,src,dst,NE,NN,HD1,EB,NM);
  fill_u32_k<<<cdiv((long long)NN*HD1,T),T,0,stream>>>((unsigned*)NS,0u,(long long)NN*HD1);
  edge_exp_k<<<cdiv(ET*HD1,T),T,0,stream>>>(EB,NM,dst,NE,NN,HD1,NS);
  rows_bias_k<<<cdiv((long long)NN*HID*HD1,T),T,0,stream>>>(B,b1,(long long)NN*HID*HD1,HID*HD1);
  edge_scatter_k<<<cdiv(ET*HD1,T),T,0,stream>>>(A,EB,NS,src,dst,NE,NN,HD1,B);
  STATS(B, NN, HID*HD1, g1, be1);
  bn_lrelu_k<<<cdiv((long long)NN*HID*HD1,T),T,0,stream>>>(B,SCL,SHF,(long long)NN*HID*HD1,HID*HD1,0.01f);

  // ---- GAT layer 2 (4 heads) ----
  GEMM(B, W2, (const float*)nullptr, A, NN, HID*HD2, HID*HD1, 0);       // hh2 (reuse A)
  attn_logits_k<<<cdiv((long long)NN*HD2,T),T,0,stream>>>(A,a2s,a2d,ALS,ALD,NN,HD2);
  fill_u32_k<<<cdiv((long long)NN*HD2,T),T,0,stream>>>(NM,ENC_NEG_FLT_MAX,(long long)NN*HD2);
  edge_max_k<<<cdiv(ET*HD2,T),T,0,stream>>>(ALS,ALD,src,dst,NE,NN,HD2,EB,NM);
  fill_u32_k<<<cdiv((long long)NN*HD2,T),T,0,stream>>>((unsigned*)NS,0u,(long long)NN*HD2);
  edge_exp_k<<<cdiv(ET*HD2,T),T,0,stream>>>(EB,NM,dst,NE,NN,HD2,NS);
  rows_bias_k<<<cdiv((long long)NN*HID*HD2,T),T,0,stream>>>(Cb,b2,(long long)NN*HID*HD2,HID*HD2);
  edge_scatter_k<<<cdiv(ET*HD2,T),T,0,stream>>>(A,EB,NS,src,dst,NE,NN,HD2,Cb);
  STATS(Cb, NN, HID*HD2, g2, be2);
  bn_lrelu_k<<<cdiv((long long)NN*HID*HD2,T),T,0,stream>>>(Cb,SCL,SHF,(long long)NN*HID*HD2,HID*HD2,0.01f); // xg2

  // ---- GCN branch ----
  GEMM(H, Wg, (const float*)nullptr, B, NN, 4*HID, HID, 0);             // hwg (reuse B)
  fill_u32_k<<<cdiv((long long)NN,T),T,0,stream>>>((unsigned*)DEG,0x3F800000u,NN); // deg=1 (self-loop)
  deg_k<<<cdiv((long long)NE,T),T,0,stream>>>(dst,DEG,NE);
  dis_k<<<cdiv((long long)NN,T),T,0,stream>>>(DEG,NN);
  gcn_init_k<<<cdiv((long long)NN*4*HID,T),T,0,stream>>>(A,B,DEG,bg,(long long)NN*4*HID,4*HID);
  gcn_scatter_k<<<cdiv((long long)NE,T),T,0,stream>>>(B,DEG,src,dst,A,NE,4*HID);
  STATS(A, NN, 4*HID, gg, beg);
  bn_lrelu_k<<<cdiv((long long)NN*4*HID,T),T,0,stream>>>(A,SCL,SHF,(long long)NN*4*HID,4*HID,0.01f); // xgc

  // ---- gated fusion: gate = sigmoid([xg2,xgc] @ Wgate + bgate); xc -> B ----
  GEMM(Cb, Wgate,                  bgate,                  B, NN, 4*HID, 4*HID, 0);
  GEMM(A,  Wgate + (size_t)512*512,(const float*)nullptr, B, NN, 4*HID, 4*HID, 1);
  gate_combine_k<<<cdiv((long long)NN*4*HID,T),T,0,stream>>>(B,Cb,A,(long long)NN*4*HID);

  // ---- attention pooling ----
  GEMM(B, Wp1, bp1, A, NN, HID, 4*HID, 0);                              // [N,128] into A
  lrelu_k<<<cdiv((long long)NN*HID,T),T,0,stream>>>(A,(long long)NN*HID,0.2f);
  score_k<<<cdiv((long long)NN,T),T,0,stream>>>(A,Wp2,bp2,SC,NN);
  fill_u32_k<<<cdiv((long long)NG,T),T,0,stream>>>(GM,ENC_NEG_FLT_MAX,NG);
  seg_max_k<<<cdiv((long long)NN,T),T,0,stream>>>(SC,bat,GM,NN);
  fill_u32_k<<<cdiv((long long)NG,T),T,0,stream>>>((unsigned*)GS,0u,NG);
  seg_exp_k<<<cdiv((long long)NN,T),T,0,stream>>>(SC,bat,GM,GS,NN);
  fill_u32_k<<<cdiv((long long)NG*512,T),T,0,stream>>>((unsigned*)PL,0u,(long long)NG*512);
  pool_scatter_k<<<cdiv((long long)NN,T),T,0,stream>>>(SC,GS,bat,B,PL,NN,4*HID);

  // ---- classifier ----
  GEMM(PL, Wc1, bc1, HC, NG, HID, 4*HID, 0);
  STATS(HC, NG, HID, gc, bec);
  bn_lrelu_k<<<cdiv((long long)NG*HID,T),T,0,stream>>>(HC,SCL,SHF,(long long)NG*HID,HID,0.0f); // ReLU
  GEMM(HC, Wc2, bc2, LG, NG, OUTC, HID, 0);
  log_softmax_k<<<cdiv((long long)NG,T),T,0,stream>>>(LG,(float*)d_out,NG,OUTC);

  #undef GEMM
  #undef STATS
}
